// XFMultiHeadAttention_661424963632
// MI455X (gfx1250) — matrix operand order repaired and checked by reading, not compiled, not run on hardware
//
#include <hip/hip_runtime.h>

#define BB 4
#define SS 2048
#define DD 1024
#define HH 16
#define DEPTH 64

typedef __attribute__((ext_vector_type(16))) __bf16 v16bf;
typedef __attribute__((ext_vector_type(8)))  __bf16 v8bf;
typedef __attribute__((ext_vector_type(8)))  float  v8f;
typedef __attribute__((ext_vector_type(4)))  unsigned int u32x4;
typedef __attribute__((ext_vector_type(8)))  int i32x8;
typedef __attribute__((ext_vector_type(4)))  int i32x4;

union Frag { v16bf v; v8bf h[2]; };

__device__ inline unsigned short f2bf(float f) {
  unsigned int u = __builtin_bit_cast(unsigned int, f);
  u += 0x7FFFu + ((u >> 16) & 1u);        // round-to-nearest-even
  return (unsigned short)(u >> 16);
}

__device__ inline v8f zero8() {
  v8f z = {0.f, 0.f, 0.f, 0.f, 0.f, 0.f, 0.f, 0.f};
  return z;
}

__device__ inline v8f wmma_bf16(const Frag& a, const Frag& b, v8f c) {
  return __builtin_amdgcn_wmma_f32_16x16x32_bf16(
      false, a.v, false, b.v, (short)0, c, false, false);
}

// Low 32 bits of a flat pointer to __shared__ = LDS byte address.
__device__ inline unsigned lds_off(const void* p) {
  return (unsigned)(unsigned long long)(uintptr_t)p;
}

// ---------------------------------------------------------------------------
// Tensor Data Mover: issue a 2-D tile load (bf16 elements) into LDS.
// D# layout per CDNA5 ISA (groups 2/3 zero: 2-D tile,
// workgroup_mask = 0 since we are not dispatched in a cluster).
// This toolchain exposes the 6-arg builtin: (g0, g1, g2, g3, g4, cpol).
//   tile_d0 = contiguous elements per row, tile_d1 = rows,
//   stride0 = row stride in elements.
// ---------------------------------------------------------------------------
__device__ inline void tdm_load_2d_bf16(unsigned ldsaddr, const void* gptr,
                                        unsigned tile_d0, unsigned tile_d1,
                                        unsigned stride0) {
  unsigned long long ga = (unsigned long long)(uintptr_t)gptr;
  u32x4 g0;
  g0[0] = 1u;                                        // count=1, user descriptor
  g0[1] = ldsaddr;                                   // lds_addr [63:32]
  g0[2] = (unsigned)(ga & 0xffffffffu);              // global_addr [95:64]
  g0[3] = (unsigned)((ga >> 32) & 0x1ffffffu)        // global_addr [120:96]
          | (2u << 30);                              // type=2 ("image")
  i32x8 g1;
  g1[0] = (int)(1u << 16);                           // data_size=1 -> 2 bytes
  g1[1] = (int)((tile_d0 & 0xffffu) << 16);          // tensor_dim0[15:0]
  g1[2] = (int)(((tile_d0 >> 16) & 0xffffu)          // tensor_dim0[31:16]
          | ((tile_d1 & 0xffffu) << 16));            // tensor_dim1[15:0]
  g1[3] = (int)(((tile_d1 >> 16) & 0xffffu)          // tensor_dim1[31:16]
          | ((tile_d0 & 0xffffu) << 16));            // tile_dim0
  g1[4] = (int)(tile_d1 & 0xffffu);                  // tile_dim1 (tile_dim2=0)
  g1[5] = (int)stride0;                              // tensor_dim0_stride[31:0]
  g1[6] = 0;                                         // stride0 hi, stride1 lo
  g1[7] = 0;
  i32x4 z4 = {0, 0, 0, 0};
  i32x8 z8 = {0, 0, 0, 0, 0, 0, 0, 0};
  __builtin_amdgcn_tensor_load_to_lds(g0, g1, z4, z4, z8, 0);
}

// ---------------------------------------------------------------------------
// Kernel 0: convert f32 weights -> bf16, transposed to [N][K] layout so the
// GEMM B-tile stages straight into the WMMA B-fragment striping.
// ---------------------------------------------------------------------------
__global__ __launch_bounds__(256)
void prep_w(const float* __restrict__ Wq, const float* __restrict__ Wk,
            const float* __restrict__ Wv, const float* __restrict__ Wo,
            unsigned short* __restrict__ wt) {
  const int w = blockIdx.y;
  const float* src = (w == 0) ? Wq : (w == 1) ? Wk : (w == 2) ? Wv : Wo;
  unsigned short* dst = wt + (size_t)w * DD * DD;
  const int i = blockIdx.x * 256 + threadIdx.x;   // write index, coalesced
  const int n = i / DD, kk = i % DD;
  dst[i] = f2bf(src[(size_t)kk * DD + n]);        // Wt[n][k] = W[k][n]
}

// ---------------------------------------------------------------------------
// Tiled GEMM: C[M=8192, N=1024] = A[M,1024] * Wt^T + bias.
// Block tile 128x128, 256 threads (8 waves), wave tile 32x64 (2x4 frags).
// B tile is staged by the Tensor Data Mover (wave 0 issues, all waves overlap
// the A-tile f32->bf16 conversion with the DMA).
// AF32: A is f32 (converted to bf16 while staging), else A is bf16.
// MODE 0: write bf16 head-split [B,H,S,64]; MODE 1: bf16 Vt [B,H,64,S];
// MODE 2: write f32 row-major (final output).
// ---------------------------------------------------------------------------
template <bool AF32, int MODE>
__global__ __launch_bounds__(256)
void gemm_bf16_k(const void* __restrict__ Aptr,
                 const unsigned short* __restrict__ Bt,
                 const float* __restrict__ bias, void* __restrict__ Optr) {
  constexpr int Kd = DD, Nd = DD;
  const int m0 = blockIdx.y * 128;
  const int n0 = blockIdx.x * 128;
  const int tid = threadIdx.x;
  const int wave = tid >> 5, lane = tid & 31;
  const int lane16 = lane & 15, hf = lane >> 4;
  const int wm = (wave & 3) * 32;   // wave M offset in block tile
  const int wn = (wave >> 2) * 64;  // wave N offset in block tile

  __shared__ __align__(16) unsigned short As[128 * 32];  // [m][k] bf16
  __shared__ __align__(16) unsigned short Bs[128 * 32];  // [n][k] bf16 (B^T)

  v8f acc[2][4];
#pragma unroll
  for (int im = 0; im < 2; ++im)
#pragma unroll
    for (int in = 0; in < 4; ++in) acc[im][in] = zero8();

  const unsigned bs_lds = lds_off(&Bs[0]);

  for (int k0 = 0; k0 < Kd; k0 += 32) {
    // ---- B tile via TDM: 128 rows x 32 k, row stride Kd (issued async) ----
    if (wave == 0)
      tdm_load_2d_bf16(bs_lds, Bt + (size_t)n0 * Kd + k0, 32, 128, Kd);

    // ---- stage A tile (convert f32->bf16 on the fly when AF32) ----
    if (AF32) {
      const float* A = (const float*)Aptr;
#pragma unroll
      for (int i = 0; i < 4; ++i) {
        int idx = tid + i * 256;             // 1024 quads of 4 floats
        int row = idx >> 3, cg = idx & 7;
        float4 x = *(const float4*)(A + (size_t)(m0 + row) * Kd + k0 + cg * 4);
        unsigned int p0 = (unsigned int)f2bf(x.x) | ((unsigned int)f2bf(x.y) << 16);
        unsigned int p1 = (unsigned int)f2bf(x.z) | ((unsigned int)f2bf(x.w) << 16);
        uint2 pk; pk.x = p0; pk.y = p1;
        *(uint2*)&As[row * 32 + cg * 4] = pk;
      }
      if (k0 + 32 < Kd)   // L2 prefetch of next A tile (global_prefetch_b8)
        __builtin_prefetch(A + (size_t)(m0 + (tid >> 3)) * Kd + (k0 + 32) +
                               (tid & 7) * 4, 0, 1);
    } else {
      const unsigned short* A = (const unsigned short*)Aptr;
#pragma unroll
      for (int i = 0; i < 2; ++i) {
        int idx = tid + i * 256;             // 512 chunks of 8 bf16
        int row = idx >> 2, seg = idx & 3;
        *(uint4*)&As[row * 32 + seg * 8] =
            *(const uint4*)(A + (size_t)(m0 + row) * Kd + k0 + seg * 8);
      }
    }
    if (wave == 0) __builtin_amdgcn_s_wait_tensorcnt(0);
    __syncthreads();

    // ---- fragments + 8 WMMA per K-step ----
    Frag a[2];
#pragma unroll
    for (int im = 0; im < 2; ++im) {
      const unsigned short* base = &As[(wm + im * 16 + lane16) * 32];
      a[im].h[0] = *(const v8bf*)(base + hf * 8);        // K = hf*8 .. +7
      a[im].h[1] = *(const v8bf*)(base + 16 + hf * 8);   // K = 16+hf*8 .. +7
    }
#pragma unroll
    for (int in = 0; in < 4; ++in) {
      Frag b;
      const unsigned short* base = &Bs[(wn + in * 16 + lane16) * 32 + hf * 8];
      b.h[0] = *(const v8bf*)(base);                     // K = hf*8 .. +7
      b.h[1] = *(const v8bf*)(base + 16);                // K = 16+hf*8 .. +7
#pragma unroll
      for (int im = 0; im < 2; ++im)
        acc[im][in] = wmma_bf16(a[im], b, acc[im][in]);
    }
    __syncthreads();
  }

  // ---- epilogue: bias + layout-specific store ----
#pragma unroll
  for (int im = 0; im < 2; ++im) {
#pragma unroll
    for (int in = 0; in < 4; ++in) {
      int gn = n0 + wn + in * 16 + lane16;
      float bv = bias[gn];
#pragma unroll
      for (int r = 0; r < 8; ++r) {
        int gm = m0 + wm + im * 16 + r + 8 * hf;  // C layout: row = r + 8*half
        float val = acc[im][in][r] + bv;
        if (MODE == 2) {
          ((float*)Optr)[(size_t)gm * Nd + gn] = val;
        } else {
          int bb = gm >> 11, ss = gm & (SS - 1);
          int hh = gn >> 6, dd = gn & 63;
          unsigned short* O = (unsigned short*)Optr;
          if (MODE == 0)
            O[(((size_t)bb * HH + hh) * SS + ss) * DEPTH + dd] = f2bf(val);
          else
            O[(((size_t)bb * HH + hh) * DEPTH + dd) * SS + ss] = f2bf(val);
        }
      }
    }
  }
}

// ---------------------------------------------------------------------------
// Flash attention: one block = (b, h, 64 q-rows); 4 waves x 16 q-rows.
// Online softmax over KV chunks of 32; K/V tiles DMA'd into LDS by the TDM.
// ---------------------------------------------------------------------------
__global__ __launch_bounds__(128)
void flash_attn(const unsigned short* __restrict__ Qh,
                const unsigned short* __restrict__ Kh,
                const unsigned short* __restrict__ Vt,
                const float* __restrict__ mask,
                unsigned short* __restrict__ Obuf) {
  const int q0 = blockIdx.x * 64, h = blockIdx.y, b = blockIdx.z;
  const int tid = threadIdx.x, wave = tid >> 5, lane = tid & 31;
  const int lane16 = lane & 15, hf = lane >> 4;
  const size_t bh = (size_t)(b * HH + h);

  __shared__ __align__(16) unsigned short Kt[32 * 64];      // [kvrow][depth]
  __shared__ __align__(16) unsigned short Vs[64 * 32];      // [depth][kvrow] (B^T)
  __shared__ __align__(16) unsigned short Pb[4 * 16 * 32];  // per-wave P tile

  // Q fragments (16 rows x 64 depth = 2 K-steps), loaded once from global.
  const unsigned short* Qbase = Qh + (bh * SS + q0 + wave * 16 + lane16) * DEPTH;
  Frag qf[2];
#pragma unroll
  for (int ks = 0; ks < 2; ++ks) {
    qf[ks].h[0] = *(const v8bf*)(Qbase + ks * 32 + hf * 8);
    qf[ks].h[1] = *(const v8bf*)(Qbase + ks * 32 + 16 + hf * 8);
  }

  v8f of[4];
#pragma unroll
  for (int f = 0; f < 4; ++f) of[f] = zero8();
  float mrow[8], lrow[8];
#pragma unroll
  for (int r = 0; r < 8; ++r) { mrow[r] = -1.0e30f; lrow[r] = 0.f; }

  unsigned short* Pw = &Pb[wave * 512];
  const unsigned kt_lds = lds_off(&Kt[0]);
  const unsigned vs_lds = lds_off(&Vs[0]);
  const unsigned short* Kg = Kh + bh * SS * DEPTH;
  const unsigned short* Vg = Vt + bh * DEPTH * (size_t)SS;

  for (int kv = 0; kv < SS; kv += 32) {
    // ---- TDM: K chunk [32][64] (stride 64) and V^T chunk [64][32] (stride S)
    if (wave == 0) {
      tdm_load_2d_bf16(kt_lds, Kg + (size_t)kv * DEPTH, 64, 32, DEPTH);
      tdm_load_2d_bf16(vs_lds, Vg + kv, 32, 64, SS);
      __builtin_amdgcn_s_wait_tensorcnt(0);
    }
    __syncthreads();

    // ---- S = Q K^T / sqrt(d) + mask  (2 N-frags, K=64 in 2 steps) ----
    v8f sf[2];
#pragma unroll
    for (int f = 0; f < 2; ++f) {
      v8f acc = zero8();
#pragma unroll
      for (int ks = 0; ks < 2; ++ks) {
        Frag bfr;
        const unsigned short* base = &Kt[(f * 16 + lane16) * 64 + ks * 32 + hf * 8];
        bfr.h[0] = *(const v8bf*)(base);
        bfr.h[1] = *(const v8bf*)(base + 16);
        acc = wmma_bf16(qf[ks], bfr, acc);
      }
      float mv = mask[(size_t)b * SS + kv + f * 16 + lane16] * (-1e9f);
#pragma unroll
      for (int r = 0; r < 8; ++r) sf[f][r] = acc[r] * 0.125f + mv;
    }

    // ---- online softmax (row = r + 8*hf lives across 16 lanes) ----
    float mnew[8];
#pragma unroll
    for (int r = 0; r < 8; ++r) {
      float cm = fmaxf(sf[0][r], sf[1][r]);
      cm = fmaxf(cm, __shfl_xor(cm, 1, 32));
      cm = fmaxf(cm, __shfl_xor(cm, 2, 32));
      cm = fmaxf(cm, __shfl_xor(cm, 4, 32));
      cm = fmaxf(cm, __shfl_xor(cm, 8, 32));
      mnew[r] = fmaxf(mrow[r], cm);
    }
#pragma unroll
    for (int r = 0; r < 8; ++r) {
      float alpha = __expf(mrow[r] - mnew[r]);
      lrow[r] *= alpha;
#pragma unroll
      for (int f = 0; f < 4; ++f) of[f][r] *= alpha;
      float p0 = __expf(sf[0][r] - mnew[r]);
      float p1 = __expf(sf[1][r] - mnew[r]);
      sf[0][r] = p0; sf[1][r] = p1;
      float ps = p0 + p1;
      ps += __shfl_xor(ps, 1, 32);
      ps += __shfl_xor(ps, 2, 32);
      ps += __shfl_xor(ps, 4, 32);
      ps += __shfl_xor(ps, 8, 32);
      lrow[r] += ps;
      mrow[r] = mnew[r];
    }

    // ---- C-layout P -> A-layout via wave-private LDS bounce ----
#pragma unroll
    for (int f = 0; f < 2; ++f)
#pragma unroll
      for (int r = 0; r < 8; ++r)
        Pw[(r + 8 * hf) * 32 + f * 16 + lane16] = f2bf(sf[f][r]);
    asm volatile("s_wait_dscnt 0x0" ::: "memory");  // wave-local LDS RAW fence
    Frag pf;
    pf.h[0] = *(const v8bf*)&Pw[lane16 * 32 + hf * 8];
    pf.h[1] = *(const v8bf*)&Pw[lane16 * 32 + 16 + hf * 8];

    // ---- O += P V  (4 depth frags) ----
#pragma unroll
    for (int f = 0; f < 4; ++f) {
      Frag vf;
      const unsigned short* base = &Vs[(f * 16 + lane16) * 32 + hf * 8];
      vf.h[0] = *(const v8bf*)(base);
      vf.h[1] = *(const v8bf*)(base + 16);
      of[f] = wmma_bf16(pf, vf, of[f]);
    }
    __syncthreads();
  }

  // ---- normalize and store merged-head O (bf16, [B,S,D]) ----
#pragma unroll
  for (int r = 0; r < 8; ++r) lrow[r] = 1.0f / lrow[r];
#pragma unroll
  for (int f = 0; f < 4; ++f)
#pragma unroll
    for (int r = 0; r < 8; ++r) {
      int s = q0 + wave * 16 + r + 8 * hf;
      Obuf[((size_t)b * SS + s) * DD + h * DEPTH + f * 16 + lane16] =
          f2bf(of[f][r] * lrow[r]);
    }
}

// ---------------------------------------------------------------------------
extern "C" void kernel_launch(void* const* d_in, const int* in_sizes, int n_in,
                              void* d_out, int out_size, void* d_ws, size_t ws_size,
                              hipStream_t stream) {
  (void)in_sizes; (void)n_in; (void)out_size; (void)ws_size;
  const float* v    = (const float*)d_in[0];
  const float* k    = (const float*)d_in[1];
  const float* q    = (const float*)d_in[2];
  const float* mask = (const float*)d_in[3];
  const float* Wq   = (const float*)d_in[4];
  const float* bq   = (const float*)d_in[5];
  const float* Wk   = (const float*)d_in[6];
  const float* bk   = (const float*)d_in[7];
  const float* Wv   = (const float*)d_in[8];
  const float* bv   = (const float*)d_in[9];
  const float* Wo   = (const float*)d_in[10];
  const float* bo   = (const float*)d_in[11];

  unsigned short* ws = (unsigned short*)d_ws;
  const size_t WSZ = (size_t)DD * DD;        // 1M elems per weight
  const size_t TSZ = (size_t)BB * SS * DD;   // 8M elems per tensor
  unsigned short* wt = ws;                   // 4 transposed bf16 weights
  unsigned short* Qh = ws + 4 * WSZ;         // [B,H,S,64]
  unsigned short* Kh = Qh + TSZ;             // [B,H,S,64]
  unsigned short* Vt = Kh + TSZ;             // [B,H,64,S]
  unsigned short* Ob = Vt + TSZ;             // [B,S,D]

  prep_w<<<dim3(DD * DD / 256, 4), 256, 0, stream>>>(Wq, Wk, Wv, Wo, wt);

  dim3 ggrid(DD / 128, BB * SS / 128);
  gemm_bf16_k<true, 0><<<ggrid, 256, 0, stream>>>(q, wt + 0 * WSZ, bq, Qh);
  gemm_bf16_k<true, 0><<<ggrid, 256, 0, stream>>>(k, wt + 1 * WSZ, bk, Kh);
  gemm_bf16_k<true, 1><<<ggrid, 256, 0, stream>>>(v, wt + 2 * WSZ, bv, Vt);

  flash_attn<<<dim3(SS / 64, HH, BB), 128, 0, stream>>>(Qh, Kh, Vt, mask, Ob);

  gemm_bf16_k<false, 2><<<ggrid, 256, 0, stream>>>(Ob, wt + 3 * WSZ, bo, d_out);
}
